// DeformCapsNet_37615323578636
// MI455X (gfx1250) — matrix operand order repaired and checked by reading, not compiled, not run on hardware
//
#include <hip/hip_runtime.h>
#include <cstddef>
#include <cstdint>

// ---------- types for WMMA ----------
typedef __attribute__((ext_vector_type(16))) __bf16 bf16x16;
typedef __attribute__((ext_vector_type(8)))  float  f32x8;

struct Pack32 { uint4 a, b; };   // 32 bytes, bit-castable to bf16x16

__device__ __forceinline__ unsigned short f2bf(float f) {
    unsigned u = __builtin_bit_cast(unsigned, f);
    u += 0x7FFFu + ((u >> 16) & 1u);           // round-to-nearest-even
    return (unsigned short)(u >> 16);
}
__device__ __forceinline__ float bf2f(unsigned short h) {
    return __builtin_bit_cast(float, ((unsigned)h) << 16);
}

// LDS byte address (low 32 bits of a generic pointer into shared memory)
__device__ __forceinline__ unsigned lds_addr(const void* p) {
    return (unsigned)(unsigned long long)p;
}
// CDNA5 async global->LDS copy, GV mode (ISA 15.18.3 opc 98). Tracked by ASYNCcnt.
__device__ __forceinline__ void async_ld_b128(unsigned lds, unsigned long long gaddr) {
    asm volatile("global_load_async_to_lds_b128 %0, %1, off"
                 :: "v"(lds), "v"(gaddr) : "memory");
}
template<int N>
__device__ __forceinline__ void wait_asynccnt() {
    asm volatile("s_wait_asynccnt %0" :: "n"(N) : "memory");
}

static constexpr int Bn = 8, Hh = 96, Ww = 96, HW = Hh * Ww;   // 9216
static constexpr int OC = 8;                                   // votes channels = 1024

// ---------------------------------------------------------------------------
// weight convert: src f32 [COUT][CIN][3][3]  ->  dst bf16 [COUT][9*CIN] (k = r*CIN+ci)
// ---------------------------------------------------------------------------
template<int CIN, int COUT>
__global__ void cvt_w_kernel(const float* __restrict__ src, unsigned short* __restrict__ dst) {
    const int K = 9 * CIN;
    int id = blockIdx.x * 256 + threadIdx.x;
    if (id >= COUT * K) return;
    int n = id / K, k = id - n * K;
    int r = k / CIN, ci = k - r * CIN;
    dst[id] = f2bf(src[(n * CIN + ci) * 9 + r]);
}

// ---------------------------------------------------------------------------
// conv1: 3 -> 64, pad 1, ReLU, NCHW f32 in -> NHWC bf16 out
// ---------------------------------------------------------------------------
__global__ void conv1_kernel(const float* __restrict__ x, const float* __restrict__ w,
                             const float* __restrict__ b, unsigned short* __restrict__ f1) {
    int id = blockIdx.x * 256 + threadIdx.x;            // = pos*64 + co
    if (id >= Bn * HW * 64) return;
    int co = id & 63;
    int pos = id >> 6;
    int px = pos % Ww;
    int rem = pos / Ww;
    int py = rem % Hh;
    int pb = rem / Hh;
    float acc = b[co];
    #pragma unroll
    for (int ci = 0; ci < 3; ++ci)
        #pragma unroll
        for (int dy = 0; dy < 3; ++dy)
            #pragma unroll
            for (int dx = 0; dx < 3; ++dx) {
                int yy = py + dy - 1, xx = px + dx - 1;
                if ((unsigned)yy < (unsigned)Hh && (unsigned)xx < (unsigned)Ww)
                    acc += x[((pb * 3 + ci) * Hh + yy) * Ww + xx] *
                           w[((co * 3 + ci) * 3 + dy) * 3 + dx];
            }
    f1[id] = f2bf(fmaxf(acc, 0.f));
}

// ---------------------------------------------------------------------------
// Implicit-GEMM 3x3 conv via WMMA bf16, double-buffered LDS with async copies.
//   fin : NHWC bf16 [B,96,96,CIN]
//   wT  : bf16 [COUT][9*CIN]  (k = r*CIN+ci)
//   MODE 0: out = ReLU(conv+bias) -> NHWC bf16 [B,96,96,COUT]
//   MODE 1: out = conv+bias -> votes bf16 [B, O=8, HW, 128]
// Block: 128 threads = 4 waves (2x2), block tile 64(M) x 128(N),
// wave tile 32(M) x 64(N) = 8 WMMA accumulators, K step 32.
// ---------------------------------------------------------------------------
template<int CIN, int COUT, int MODE>
__global__ void wmma_conv_kernel(const unsigned short* __restrict__ fin,
                                 const unsigned short* __restrict__ wT,
                                 const float* __restrict__ bias,
                                 unsigned short* __restrict__ out) {
    constexpr int K = 9 * CIN;
    constexpr int AST = 40;   // ushort stride (32 + 8 pad), rows stay 16B aligned
    constexpr int BST = 40;
    __shared__ __align__(16) unsigned short As[2][64 * AST];
    __shared__ __align__(16) unsigned short Bs[2][128 * BST];
    __shared__ __align__(16) unsigned short dumpS[8];   // sink for OOB async lanes

    const int t = threadIdx.x;             // 0..127
    const int lane = t & 31;
    const int wave = t >> 5;               // 0..3
    const int waveM = wave >> 1;           // 0..1
    const int waveN = wave & 1;            // 0..1
    const int blockM = blockIdx.x;         // M/64
    const int blockN = blockIdx.y;         // COUT/128

    // A staging: thread -> (row, 16-element half-row)
    const int arow = t >> 1;               // 0..63
    const int aseg = (t & 1) * 16;         // 0 or 16 (ushorts)
    const int am = blockM * 64 + arow;
    const int ab = am / HW;
    const int ahw = am - ab * HW;
    const int ay = ahw / Ww;
    const int ax = ahw - ay * Ww;
    // B staging: thread -> one row (64B)
    const int brow = t;                    // 0..127

    const unsigned dumpA = lds_addr(&dumpS[0]);

    // issue one K-stage of async copies into buffer `buf`
    auto stage = [&](int kk, int buf) {
        int r = kk / CIN;
        int cb = (kk - r * CIN) + aseg;
        int dy = r / 3 - 1, dx = r - (r / 3) * 3 - 1;
        int yy = ay + dy, xx = ax + dx;
        bool inb = ((unsigned)yy < (unsigned)Hh) && ((unsigned)xx < (unsigned)Ww);
        int py = inb ? yy : ay, px = inb ? xx : ax;     // clamped (always valid) source
        unsigned long long ga =
            (unsigned long long)(fin + ((size_t)(ab * HW + py * Ww + px) * CIN + cb));
        unsigned short* lreal = &As[buf][arow * AST + aseg];
        unsigned la0 = lds_addr(lreal), la1 = lds_addr(lreal + 8);
        if (!inb) {
            uint4 z = make_uint4(0u, 0u, 0u, 0u);
            *(uint4*)(lreal) = z;
            *(uint4*)(lreal + 8) = z;
            la0 = dumpA;
            la1 = dumpA;
        }
        async_ld_b128(la0, ga);                 // 1
        async_ld_b128(la1, ga + 16);            // 2
        const unsigned short* bsrc = wT + (size_t)(blockN * 128 + brow) * K + kk;
        unsigned short* bl = &Bs[buf][brow * BST];
        #pragma unroll
        for (int s = 0; s < 4; ++s)             // 3..6
            async_ld_b128(lds_addr(bl + s * 8), (unsigned long long)(bsrc + s * 8));
    };

    f32x8 acc[2][4];
    #pragma unroll
    for (int mi = 0; mi < 2; ++mi)
        #pragma unroll
        for (int ni = 0; ni < 4; ++ni)
            #pragma unroll
            for (int j = 0; j < 8; ++j) acc[mi][ni][j] = 0.f;

    const int mrow = lane & 15;
    const int hi = lane >> 4;

    stage(0, 0);                                   // prologue
    int cur = 0;
    for (int kk = 0; kk < K; kk += 32, cur ^= 1) {
        if (kk + 32 < K) {
            stage(kk + 32, cur ^ 1);               // overlap next stage with this compute
            wait_asynccnt<6>();                    // in-order => stage(kk) complete
        } else {
            wait_asynccnt<0>();
        }
        __syncthreads();                           // all waves' tiles visible

        bf16x16 af[2], bf[4];
        #pragma unroll
        for (int mi = 0; mi < 2; ++mi) {
            int ml = waveM * 32 + mi * 16 + mrow;
            Pack32 p;
            p.a = *(const uint4*)(&As[cur][ml * AST + hi * 8]);       // K = kbase..+7
            p.b = *(const uint4*)(&As[cur][ml * AST + 16 + hi * 8]);  // K = 16+kbase..
            af[mi] = __builtin_bit_cast(bf16x16, p);
        }
        #pragma unroll
        for (int ni = 0; ni < 4; ++ni) {
            int nl = waveN * 64 + ni * 16 + mrow;
            Pack32 p;
            p.a = *(const uint4*)(&Bs[cur][nl * BST + hi * 8]);
            p.b = *(const uint4*)(&Bs[cur][nl * BST + 16 + hi * 8]);
            bf[ni] = __builtin_bit_cast(bf16x16, p);
        }
        #pragma unroll
        for (int mi = 0; mi < 2; ++mi)
            #pragma unroll
            for (int ni = 0; ni < 4; ++ni)
                acc[mi][ni] = __builtin_amdgcn_wmma_f32_16x16x32_bf16(
                    false, af[mi], false, bf[ni],
                    (short)0, acc[mi][ni], false, false);

        __syncthreads();                           // readers done before tile is re-staged
    }

    // ---- epilogue ----
    #pragma unroll
    for (int mi = 0; mi < 2; ++mi)
        #pragma unroll
        for (int ni = 0; ni < 4; ++ni)
            #pragma unroll
            for (int j = 0; j < 8; ++j) {
                int m = blockM * 64 + waveM * 32 + mi * 16 + j + 8 * hi;
                int n = blockN * 128 + waveN * 64 + ni * 16 + (lane & 15);
                float v = acc[mi][ni][j] + bias[n];
                if (MODE == 0) {
                    v = fmaxf(v, 0.f);
                    out[(size_t)m * COUT + n] = f2bf(v);
                } else {
                    int o = n >> 7, rest = n & 127;
                    int b = m / HW, hw = m - b * HW;
                    out[((size_t)(b * OC + o) * HW + hw) * 128 + rest] = f2bf(v);
                }
            }
}

// ---------------------------------------------------------------------------
// zero
// ---------------------------------------------------------------------------
__global__ void zerof_kernel(float* __restrict__ p, int n) {
    int i = blockIdx.x * 256 + threadIdx.x;
    if (i < n) p[i] = 0.f;
}

// ---------------------------------------------------------------------------
// One routing iteration, fused:
//  phase1 (thread = hw in 128-chunk): agree -> logits update -> softmax(route)
//  phase2 (thread = (c,a)): partial preact over chunk, one global atomicAdd
// votes: bf16 [64(bo)][9216][128]; logits: f32 [64][9216][8]; act/preact: f32 [64][128]
// ---------------------------------------------------------------------------
__global__ void routing_iter_kernel(const unsigned short* __restrict__ votes,
                                    float* __restrict__ logits,
                                    const float* __restrict__ act_prev,
                                    float* __restrict__ preact,
                                    int iter) {
    __shared__ float routeS[128 * 8];
    const int t = threadIdx.x;              // 0..127
    const int chunk = blockIdx.x;           // 0..71
    const int bo = blockIdx.y;              // 0..63
    const int hw = chunk * 128 + t;

    const size_t vbase = ((size_t)bo * HW + hw) * 128;
    const size_t lbase = ((size_t)bo * HW + hw) * 8;

    float lg[8];
    #pragma unroll
    for (int c = 0; c < 8; ++c) lg[c] = logits[lbase + c];

    if (iter > 0) {
        #pragma unroll
        for (int c = 0; c < 8; ++c) {
            float ag = 0.f;
            #pragma unroll
            for (int a = 0; a < 16; ++a)
                ag += bf2f(votes[vbase + c * 16 + a]) * act_prev[bo * 128 + c * 16 + a];
            lg[c] += ag;
            logits[lbase + c] = lg[c];
        }
    }
    float mx = lg[0];
    #pragma unroll
    for (int c = 1; c < 8; ++c) mx = fmaxf(mx, lg[c]);
    float e[8], s = 0.f;
    #pragma unroll
    for (int c = 0; c < 8; ++c) { e[c] = __expf(lg[c] - mx); s += e[c]; }
    float inv = 1.f / s;
    #pragma unroll
    for (int c = 0; c < 8; ++c) routeS[t * 8 + c] = e[c] * inv;

    __syncthreads();

    const int c2 = t >> 4;                  // t = c*16 + a
    float acc = 0.f;
    const size_t base2 = ((size_t)bo * HW + (size_t)chunk * 128) * 128 + t;
    for (int h = 0; h < 128; ++h)
        acc += routeS[h * 8 + c2] * bf2f(votes[base2 + (size_t)h * 128]);
    atomicAdd(&preact[bo * 128 + t], acc);
}

// ---------------------------------------------------------------------------
// squash: act = (n2/(1+n2)) * v / ||v||, v = preact + caps_bias, norm over A(=16)
// ---------------------------------------------------------------------------
__global__ void squash_kernel(const float* __restrict__ preact,
                              const float* __restrict__ cbias,   // [8(o)][8(c)][16(a)]
                              float* __restrict__ act) {
    __shared__ float sq[128];
    __shared__ float nss[8];
    const int t = threadIdx.x;              // c*16+a
    const int bo = blockIdx.x;
    const int o = bo & 7;
    float v = preact[bo * 128 + t] + cbias[o * 128 + t];
    sq[t] = v * v;
    __syncthreads();
    if (t < 8) {
        float s = 0.f;
        #pragma unroll
        for (int a = 0; a < 16; ++a) s += sq[t * 16 + a];
        nss[t] = s;
    }
    __syncthreads();
    float n2 = nss[t >> 4];
    float norm = sqrtf(n2) + 1e-20f;
    act[bo * 128 + t] = (n2 / (1.f + n2)) * v / norm;
}

// ---------------------------------------------------------------------------
// SE head
// ---------------------------------------------------------------------------
__global__ void se_kernel(const float* __restrict__ instA, const float* __restrict__ clsA,
                          const float* __restrict__ w1, const float* __restrict__ b1,
                          const float* __restrict__ w2, const float* __restrict__ b2,
                          float* __restrict__ out) {
    __shared__ float comb[8 * 32];
    const int t = threadIdx.x;              // 0..255
    const int bo = blockIdx.x;              // 0..63
    const int c = t >> 5, k = t & 31;
    float cv = (k < 16) ? instA[bo * 128 + c * 16 + k]
                        : clsA[bo * 128 + c * 16 + (k - 16)];
    comb[c * 32 + k] = cv;
    __syncthreads();
    float e = b2[k];
    #pragma unroll
    for (int j = 0; j < 4; ++j) {
        float h = b1[j];
        #pragma unroll
        for (int kk = 0; kk < 32; ++kk) h += comb[c * 32 + kk] * w1[j * 32 + kk];
        h = fmaxf(h, 0.f);
        e += h * w2[k * 4 + j];
    }
    float sig = 1.f / (1.f + __expf(-e));
    out[(size_t)bo * 256 + c * 32 + k] = sig * cv;
}

// ---------------------------------------------------------------------------
// host launch
// ---------------------------------------------------------------------------
static inline char* bump(char*& p, size_t bytes) {
    char* r = p;
    p += (bytes + 255) & ~size_t(255);
    return r;
}

extern "C" void kernel_launch(void* const* d_in, const int* in_sizes, int n_in,
                              void* d_out, int out_size, void* d_ws, size_t ws_size,
                              hipStream_t stream) {
    const float* x        = (const float*)d_in[0];
    const float* conv1_w  = (const float*)d_in[1];
    const float* conv1_b  = (const float*)d_in[2];
    const float* conv2_w  = (const float*)d_in[3];
    const float* conv2_b  = (const float*)d_in[4];
    const float* inst_w   = (const float*)d_in[5];
    const float* inst_b   = (const float*)d_in[6];
    const float* inst_bias= (const float*)d_in[7];
    const float* cls_w    = (const float*)d_in[8];
    const float* cls_b    = (const float*)d_in[9];
    const float* cls_bias = (const float*)d_in[10];
    const float* se_w1    = (const float*)d_in[11];
    const float* se_b1    = (const float*)d_in[12];
    const float* se_w2    = (const float*)d_in[13];
    const float* se_b2    = (const float*)d_in[14];
    float* out = (float*)d_out;

    char* ws = (char*)d_ws;
    unsigned short* f1     = (unsigned short*)bump(ws, (size_t)Bn * HW * 64  * 2);       // 9.4 MB
    unsigned short* f2     = (unsigned short*)bump(ws, (size_t)Bn * HW * 128 * 2);       // 18.9 MB
    unsigned short* w2T    = (unsigned short*)bump(ws, (size_t)128  * 576  * 2);
    unsigned short* wInstT = (unsigned short*)bump(ws, (size_t)1024 * 1152 * 2);
    unsigned short* wClsT  = (unsigned short*)bump(ws, (size_t)1024 * 1152 * 2);
    unsigned short* votes  = (unsigned short*)bump(ws, (size_t)Bn * OC * HW * 128 * 2);  // 151 MB (reused)
    float* logits          = (float*)bump(ws, (size_t)Bn * OC * HW * 8 * 4);             // 18.9 MB
    float* preactI         = (float*)bump(ws, (size_t)3 * 64 * 128 * 4);
    float* preactC         = (float*)bump(ws, (size_t)3 * 64 * 128 * 4);
    float* actI            = (float*)bump(ws, (size_t)3 * 64 * 128 * 4);
    float* actC            = (float*)bump(ws, (size_t)3 * 64 * 128 * 4);

    // 1) weight conversion (f32 -> bf16, [N][K] transposed)
    cvt_w_kernel<64, 128><<<(128 * 576 + 255) / 256, 256, 0, stream>>>(conv2_w, w2T);
    cvt_w_kernel<128, 1024><<<(1024 * 1152 + 255) / 256, 256, 0, stream>>>(inst_w, wInstT);
    cvt_w_kernel<128, 1024><<<(1024 * 1152 + 255) / 256, 256, 0, stream>>>(cls_w, wClsT);

    // 2) backbone
    conv1_kernel<<<(Bn * HW * 64 + 255) / 256, 256, 0, stream>>>(x, conv1_w, conv1_b, f1);
    wmma_conv_kernel<64, 128, 0><<<dim3(Bn * HW / 64, 1), 128, 0, stream>>>(f1, w2T, conv2_b, f2);

    const int logitsN = Bn * OC * HW * 8;
    const int preN = 3 * 64 * 128;

    // 3) capsule layers (sequential, votes buffer reused)
    for (int layer = 0; layer < 2; ++layer) {
        const unsigned short* wT = layer ? wClsT : wInstT;
        const float* cb   = layer ? cls_b : inst_b;
        const float* bias = layer ? cls_bias : inst_bias;
        float* pre = layer ? preactC : preactI;
        float* act = layer ? actC : actI;

        zerof_kernel<<<(logitsN + 255) / 256, 256, 0, stream>>>(logits, logitsN);
        zerof_kernel<<<(preN + 255) / 256, 256, 0, stream>>>(pre, preN);

        wmma_conv_kernel<128, 1024, 1><<<dim3(Bn * HW / 64, 8), 128, 0, stream>>>(f2, wT, cb, votes);

        for (int it = 0; it < 3; ++it) {
            routing_iter_kernel<<<dim3(HW / 128, Bn * OC), 128, 0, stream>>>(
                votes, logits, it ? (act + (it - 1) * 64 * 128) : nullptr,
                pre + it * 64 * 128, it);
            squash_kernel<<<Bn * OC, 128, 0, stream>>>(pre + it * 64 * 128, bias,
                                                       act + it * 64 * 128);
        }
    }

    // 4) SE head -> output [8,8,8,32]
    se_kernel<<<Bn * OC, 256, 0, stream>>>(actI + 2 * 64 * 128, actC + 2 * 64 * 128,
                                           se_w1, se_b1, se_w2, se_b2, out);
    (void)in_sizes; (void)n_in; (void)out_size; (void)ws_size;
}
